// DTIModel_18528488915137
// MI455X (gfx1250) — matrix operand order repaired and checked by reading, not compiled, not run on hardware
//
#include <hip/hip_runtime.h>
#include <hip/hip_bf16.h>

typedef __attribute__((ext_vector_type(16))) _Float16 v16h;
typedef __attribute__((ext_vector_type(8)))  float    v8f;

// ---------------------------------------------------------------------------
// Workspace layout (floats)
//   agg     : 500000*25 = 12,500,000
//   pooled  : 16384*128 =  2,097,152
//   counts  : 16384
//   u1      : 128   (W2 @ w_fo[0:256])
//   u2      : 64    (fc_cnn_w @ w_fo[256:384])
//   consts  : 2     (b2.w_fo , fc_cnn_b.w_fo)
// ---------------------------------------------------------------------------
#define AGG_OFF    0
#define POOL_OFF   (12500000)
#define CNT_OFF    (POOL_OFF + 2097152)
#define U1_OFF     (CNT_OFF + 16384)
#define U2_OFF     (U1_OFF + 128)
#define CONST_OFF  (U2_OFF + 64)
#define WS_FLOATS  (CONST_OFF + 2)

__global__ void zero_ws_kernel(float* __restrict__ ws, int n) {
    int i = blockIdx.x * blockDim.x + threadIdx.x;
    int stride = gridDim.x * blockDim.x;
    for (; i < n; i += stride) ws[i] = 0.0f;
}

// Fold the linear tail (W2, fc_cnn_w, fc_out_w, biases) into two vectors + two scalars.
__global__ void make_fused_kernel(const float* __restrict__ W2,      // [128,256]
                                  const float* __restrict__ b2,      // [256]
                                  const float* __restrict__ fcw,     // [64,128]
                                  const float* __restrict__ fcb,     // [128]
                                  const float* __restrict__ wfo,     // [384]
                                  float* __restrict__ u1,            // [128]
                                  float* __restrict__ u2,            // [64]
                                  float* __restrict__ consts) {      // [2]
    int t = threadIdx.x;
    if (t < 128) {
        float s = 0.0f;
        for (int j = 0; j < 256; ++j) s += W2[t * 256 + j] * wfo[j];
        u1[t] = s;
    } else if (t < 192) {
        int k = t - 128;
        float s = 0.0f;
        for (int j = 0; j < 128; ++j) s += fcw[k * 128 + j] * wfo[256 + j];
        u2[k] = s;
    } else if (t == 192) {
        float s = 0.0f;
        for (int j = 0; j < 256; ++j) s += b2[j] * wfo[j];
        consts[0] = s;
    } else if (t == 193) {
        float s = 0.0f;
        for (int j = 0; j < 128; ++j) s += fcb[j] * wfo[256 + j];
        consts[1] = s;
    }
}

// Stage A: per-edge gather of x[src] (5x b128) + edge_attr, atomic scatter into agg[dst].
__global__ void edge_scatter_kernel(const float* __restrict__ x,
                                    const int*   __restrict__ ei,
                                    const float* __restrict__ ea,
                                    float* __restrict__ agg,
                                    int n_edges) {
    int e = blockIdx.x * blockDim.x + threadIdx.x;
    if (e >= n_edges) return;
    int src = ei[e];
    int dst = ei[n_edges + e];
    const float4* xr = (const float4*)(x + (long)src * 20);   // 80B rows, 16B aligned
    float* ar = agg + (long)dst * 25;
#pragma unroll
    for (int i = 0; i < 5; ++i) {
        float4 v = xr[i];
        atomicAdd(&ar[4 * i + 0], v.x);
        atomicAdd(&ar[4 * i + 1], v.y);
        atomicAdd(&ar[4 * i + 2], v.z);
        atomicAdd(&ar[4 * i + 3], v.w);
    }
    const float* er = ea + (long)e * 5;
#pragma unroll
    for (int i = 0; i < 5; ++i) atomicAdd(&ar[20 + i], er[i]);
}

__global__ void count_nodes_kernel(const int* __restrict__ batch,
                                   float* __restrict__ counts, int n_nodes) {
    int i = blockIdx.x * blockDim.x + threadIdx.x;
    if (i < n_nodes) atomicAdd(&counts[batch[i]], 1.0f);
}

// Stage B: h1 = relu(agg @ W1 + b1) via v_wmma_f32_16x16x32_f16 (K padded 25->32),
// fused with mean-pool accumulation: pooled[batch[n]] += h1[n].
// One wave per 16-node tile; 8 WMMA ops per wave (8 column tiles of 16).
// W1 staged in LDS as f16, zero-padded, TRANSPOSED [col][k] so each lane's
// 16 B K-values are one contiguous 32-byte ds_load. A padding is branchless
// (clamped index + select), so EXEC stays all-1s with no saveexec blocks.
__global__ void node_gemm_pool_wmma(const float* __restrict__ agg,   // [N,25]
                                    const float* __restrict__ W1,    // [25,128]
                                    const float* __restrict__ b1,    // [128]
                                    const int*   __restrict__ batch, // [N]
                                    float* __restrict__ pooled,      // [G,128]
                                    int n_tiles) {
    __shared__ _Float16 sW1t[128 * 32];   // [col][k], k padded to 32 -> 8 KB
    __shared__ float    sB1[128];

    // ---- cooperative LDS fill (256 threads): transpose + f32->f16 + zero-pad
    for (int idx = threadIdx.x; idx < 128 * 32; idx += blockDim.x) {
        int c = idx >> 5;          // column 0..127
        int k = idx & 31;          // K 0..31
        float v = (k < 25) ? W1[k * 128 + c] : 0.0f;
        sW1t[idx] = (_Float16)v;
    }
    if (threadIdx.x < 128) sB1[threadIdx.x] = b1[threadIdx.x];
    __syncthreads();

    int wave = (blockIdx.x * blockDim.x + threadIdx.x) >> 5;  // wave-uniform
    if (wave >= n_tiles) return;                              // whole wave exits together
    int lane = threadIdx.x & 31;
    int hi   = lane >> 4;      // lane half (0: lanes 0-15, 1: lanes 16-31)
    int mn   = lane & 15;
    int node_base = wave * 16;

    // ---- A matrix (16x32 f16): lane holds row M=mn; K-slot e maps to
    //      K = e + 16*hi. Branchless zero-pad for K>=25.
    v16h a;
    const float* arow = agg + (long)(node_base + mn) * 25;
#pragma unroll
    for (int e = 0; e < 16; ++e) {
        int k  = e + 16 * hi;
        int kc = k < 24 ? k : 24;            // clamp -> unconditional load
        float v = arow[kc];
        a[e] = (_Float16)(k < 25 ? v : 0.0f);  // v_cndmask, no branches
    }

    // graph ids for the 8 D rows this lane will scatter
    int gid[8];
#pragma unroll
    for (int r = 0; r < 8; ++r) gid[r] = batch[node_base + r + 8 * hi];

#pragma unroll
    for (int t = 0; t < 8; ++t) {
        int col = t * 16 + mn;
        // ---- B matrix (32x16 f16): lane holds col N=mn, K = e + 16*hi.
        //      One contiguous 32B LDS read (32B-aligned).
        v16h b = *(const v16h*)(&sW1t[col * 32 + 16 * hi]);

        // ---- C initialized with bias (bias depends on column only)
        float bias = sB1[col];
        v8f c;
#pragma unroll
        for (int r = 0; r < 8; ++r) c[r] = bias;

        c = __builtin_amdgcn_wmma_f32_16x16x32_f16(
                /*neg_a=*/false, a, /*neg_b=*/false, b,
                /*c_mod=*/(short)0, c, /*reuse_a=*/false, /*reuse_b=*/false);

        // ---- ReLU + scatter into pooled sums. D VGPR r -> row (r + 8*hi), col.
#pragma unroll
        for (int r = 0; r < 8; ++r) {
            float hv = c[r] > 0.0f ? c[r] : 0.0f;
            atomicAdd(&pooled[(long)gid[r] * 128 + col], hv);
        }
    }
}

// Stage C/D/E fused: per-graph conv1d(k=5)+ReLU+maxpool2 -> dot u2, plus
// mean-pooled drug features -> dot u1, sigmoid. One 64-thread block per graph.
__global__ void final_fused_kernel(const float* __restrict__ target_data, // [G,21,6]
                                   const float* __restrict__ conv_w,      // [64,21,5]
                                   const float* __restrict__ conv_b,      // [64]
                                   const float* __restrict__ pooled,      // [G,128]
                                   const float* __restrict__ counts,      // [G]
                                   const float* __restrict__ u1,          // [128]
                                   const float* __restrict__ u2,          // [64]
                                   const float* __restrict__ consts,      // [2]
                                   const float* __restrict__ fc_out_b,    // [1]
                                   float* __restrict__ out) {             // [G]
    __shared__ float s_td[21 * 6];
    __shared__ float red[64];
    int g = blockIdx.x;
    int t = threadIdx.x;

    for (int i = t; i < 126; i += 64) s_td[i] = target_data[g * 126 + i];
    __syncthreads();

    // conv output channel t at positions 0 and 1
    const float* wc = conv_w + t * 105;
    float p0 = conv_b[t], p1 = conv_b[t];
#pragma unroll
    for (int ic = 0; ic < 21; ++ic) {
        const float* row = s_td + ic * 6;
        const float* wr  = wc + ic * 5;
#pragma unroll
        for (int k = 0; k < 5; ++k) {
            p0 += wr[k] * row[k];
            p1 += wr[k] * row[k + 1];
        }
    }
    p0 = p0 > 0.0f ? p0 : 0.0f;
    p1 = p1 > 0.0f ? p1 : 0.0f;
    float y = p0 > p1 ? p0 : p1;

    float partial = y * u2[t];
    float inv = 1.0f / fmaxf(counts[g], 1.0f);
    partial += pooled[(long)g * 128 + t]      * inv * u1[t];
    partial += pooled[(long)g * 128 + t + 64] * inv * u1[t + 64];
    red[t] = partial;
    __syncthreads();

#pragma unroll
    for (int s = 32; s > 0; s >>= 1) {
        if (t < s) red[t] += red[t + s];
        __syncthreads();
    }
    if (t == 0) {
        float logit = red[0] + consts[0] + consts[1] + fc_out_b[0];
        out[g] = 1.0f / (1.0f + __expf(-logit));
    }
}

extern "C" void kernel_launch(void* const* d_in, const int* in_sizes, int n_in,
                              void* d_out, int out_size, void* d_ws, size_t ws_size,
                              hipStream_t stream) {
    const float* x           = (const float*)d_in[0];
    const int*   edge_index  = (const int*)  d_in[1];
    const float* edge_attr   = (const float*)d_in[2];
    const int*   batch       = (const int*)  d_in[3];
    const float* target_data = (const float*)d_in[4];
    const float* W1          = (const float*)d_in[5];
    const float* b1          = (const float*)d_in[6];
    const float* W2          = (const float*)d_in[7];
    const float* b2          = (const float*)d_in[8];
    const float* conv_w      = (const float*)d_in[9];
    const float* conv_b      = (const float*)d_in[10];
    const float* fc_cnn_w    = (const float*)d_in[11];
    const float* fc_cnn_b    = (const float*)d_in[12];
    const float* fc_out_w    = (const float*)d_in[13];
    const float* fc_out_b    = (const float*)d_in[14];
    float* out = (float*)d_out;

    const int n_nodes  = in_sizes[0] / 20;   // 500000
    const int n_edges  = in_sizes[1] / 2;    // 8000000
    const int n_graphs = out_size;           // 16384

    float* ws     = (float*)d_ws;
    float* agg    = ws + AGG_OFF;
    float* pooled = ws + POOL_OFF;
    float* counts = ws + CNT_OFF;
    float* u1     = ws + U1_OFF;
    float* u2     = ws + U2_OFF;
    float* consts = ws + CONST_OFF;

    // 1) zero accumulators
    zero_ws_kernel<<<2048, 256, 0, stream>>>(ws, WS_FLOATS);

    // 2) fold linear tail into u1/u2/consts
    make_fused_kernel<<<1, 256, 0, stream>>>(W2, b2, fc_cnn_w, fc_cnn_b,
                                             fc_out_w, u1, u2, consts);

    // 3) edge message + scatter-add (L2-resident agg)
    edge_scatter_kernel<<<(n_edges + 255) / 256, 256, 0, stream>>>(
        x, edge_index, edge_attr, agg, n_edges);

    // 4) per-graph node counts
    count_nodes_kernel<<<(n_nodes + 255) / 256, 256, 0, stream>>>(
        batch, counts, n_nodes);

    // 5) WMMA node GEMM (relu(agg@W1+b1)) fused with pooled accumulation
    int n_tiles = n_nodes / 16;                       // 31250, exact
    int waves_threads = n_tiles * 32;
    node_gemm_pool_wmma<<<(waves_threads + 255) / 256, 256, 0, stream>>>(
        agg, W1, b1, batch, pooled, n_tiles);

    // 6) fused CNN + head + sigmoid, one block per graph
    final_fused_kernel<<<n_graphs, 64, 0, stream>>>(
        target_data, conv_w, conv_b, pooled, counts, u1, u2, consts,
        fc_out_b, out);

    (void)n_in; (void)ws_size; (void)fc_cnn_w;
}
